// Golu_57672820850823
// MI455X (gfx1250) — matrix operand order repaired and checked by reading, not compiled, not run on hardware
//
#include <hip/hip_runtime.h>
#include <hip/hip_bf16.h>
#include <math.h>
#include <stdint.h>

// ---------------- model constants (match reference) ----------------
#define V_SZ  256
#define C_SZ  512
#define NH_SZ 8
#define HD_SZ 64          // C/NH
#define T_SZ  2048

// Use gfx1250 async global->LDS DMA (ASYNCcnt) for GEMM tile staging.
#define USE_ASYNC_STAGING 1

typedef _Float16 f16;
typedef __attribute__((ext_vector_type(16))) _Float16 v16h;
typedef __attribute__((ext_vector_type(8)))  float    v8f;

#if USE_ASYNC_STAGING
#define ASYNC_CP16(dst, src)                                            \
    asm volatile("global_load_async_to_lds_b128 %0, %1, off"            \
                 :: "v"((uint32_t)(uintptr_t)(dst)),                    \
                    "v"((uint64_t)(uintptr_t)(src))                     \
                 : "memory")
#define ASYNC_DRAIN() asm volatile("s_wait_asynccnt 0x0" ::: "memory")
#else
#define ASYNC_CP16(dst, src) (*(uint4*)(dst) = *(const uint4*)(src))
#define ASYNC_DRAIN()
#endif

// ---------------- WMMA fragment-order swizzles (ISA 7.12.2) ----------------
// A-matrix f16 16x32 (MxK): lane = (m&15) | (K-bit3 << 4); elem within lane:
//   e = ((k>>4)<<3) | (((k>>1)&3)<<1) | (k&1)   (k within 32-chunk)
// Each (16 M x 32 K) tile = 512 f16 = 1KB contiguous; tiles [m-tile][k-chunk].
__device__ __forceinline__ size_t swzA(int t, int c, int K) {
    const int cc = c & 31;
    const int e  = ((cc >> 4) << 3) | (((cc >> 1) & 3) << 1) | (cc & 1);
    const int ln = (t & 15) | (((cc >> 3) & 1) << 4);
    return ((size_t)((t >> 4) * (K >> 5) + (c >> 5)) << 9) + (ln << 4) + e;
}
// B-matrix f16 32x16 (KxN): lane = (n&15) | (K-bit3 << 4); elem = ((k>>4)<<3) | (k&7) (k within 32-chunk).
__device__ __forceinline__ size_t swzB(int n, int k, int K) {
    return ((size_t)((n >> 4) * (K >> 5) + (k >> 5)) << 9) + (((n & 15) | (((k >> 3) & 1) << 4)) << 4) + ((((k & 31) >> 4) << 3) | (k & 7));
}

// ---------------- weight f32 -> f16 swizzled conversion ----------------
__global__ __launch_bounds__(256)
void wconv_kernel(const float* __restrict__ W, f16* __restrict__ Wf, int N, int K) {
    int i = blockIdx.x * 256 + threadIdx.x;
    if (i >= N * K) return;
    int n = i / K, k = i % K;
    Wf[swzB(n, k, K)] = (f16)W[i];
}

// ---------------- embedding gather ----------------
__global__ __launch_bounds__(256)
void embed_kernel(const int* __restrict__ tokens, const float* __restrict__ emb,
                  float* __restrict__ x) {
    int i = blockIdx.x * blockDim.x + threadIdx.x;
    if (i >= T_SZ * C_SZ) return;
    int t = i / C_SZ, c = i % C_SZ;
    x[i] = emb[tokens[t] * C_SZ + c];
}

// ---------------- layernorm: block per row; writes swizzled-f16 for GEMM A ----------------
__global__ __launch_bounds__(256)
void layernorm_kernel(const float* __restrict__ x, const float* __restrict__ g,
                      const float* __restrict__ b, f16* __restrict__ yf) {
    __shared__ float red[256];
    const int row = blockIdx.x;
    const int tid = threadIdx.x;
    const float* xr = x + (size_t)row * C_SZ;

    float a0 = xr[tid], a1 = xr[tid + 256];
    red[tid] = a0 + a1;
    __syncthreads();
    for (int off = 128; off > 0; off >>= 1) {
        if (tid < off) red[tid] += red[tid + off];
        __syncthreads();
    }
    float mu = red[0] * (1.0f / C_SZ);
    __syncthreads();

    float d0 = a0 - mu, d1 = a1 - mu;
    red[tid] = d0 * d0 + d1 * d1;
    __syncthreads();
    for (int off = 128; off > 0; off >>= 1) {
        if (tid < off) red[tid] += red[tid + off];
        __syncthreads();
    }
    float rstd = rsqrtf(red[0] * (1.0f / C_SZ) + 1e-5f);

    yf[swzA(row, tid,       C_SZ)] = (f16)(d0 * rstd * g[tid]       + b[tid]);
    yf[swzA(row, tid + 256, C_SZ)] = (f16)(d1 * rstd * g[tid + 256] + b[tid + 256]);
}

// ---------------- async tile staging (branch-free, uniform strides) ----------------
__device__ __forceinline__ void stage_tiles(const f16* __restrict__ Xs,
                                            const f16* __restrict__ Ws,
                                            f16* aDst, f16* bDst,
                                            int m0, int n0, int kch, int c0, int tid) {
    {   // A: 2 m-tiles x 2 k-chunks = 4KB -> 256 x 16B units, 1 per thread
        const int mt  = tid >> 7;
        const int off = tid & 127;
        const f16* src = Xs + ((size_t)(((m0 >> 4) + mt) * kch + c0) << 9) + (off << 3);
        f16* dst = aDst + (mt << 10) + (off << 3);
        ASYNC_CP16(dst, src);
    }
    {   // B: 16 n-tiles x 2 k-chunks = 32KB -> 8 units/thread, constant stride
        const int nt0 = tid >> 7;
        const int off = tid & 127;
        const f16* src = Ws + ((size_t)(((n0 >> 4) + nt0) * kch + c0) << 9) + (off << 3);
        f16* dst = bDst + (nt0 << 10) + (off << 3);
        const size_t sstr = ((size_t)2 * kch) << 9;
#pragma unroll
        for (int j = 0; j < 8; ++j) {
            ASYNC_CP16(dst, src);
            src += sstr;
            dst += 2048;
        }
    }
}

// ---------------- WMMA GEMM: Y = act(X @ W^T + bias) ----------------
// Pre-swizzled f16 inputs. Block = 8 waves; block tile 32(M) x 256(N);
// wave tile 32x32 (four 16x16 WMMA tiles); K stepped by 64 -> 8 WMMAs/step.
// Double-buffered LDS (72KB); async DMA of next K-slab overlaps current WMMAs.
__global__ __launch_bounds__(256)
void gemm_xwt_wmma(const f16* __restrict__ Xs, const f16* __restrict__ Ws,
                   const float* __restrict__ bias, float* __restrict__ Yf32,
                   f16* __restrict__ Yf16, int M, int N, int K, int act) {
    __shared__ alignas(32) f16 aF[2][2048];
    __shared__ alignas(32) f16 bF[2][16 * 1024];

    const int tid  = threadIdx.x;
    const int wave = tid >> 5;
    const int lane = tid & 31;
    const int m0   = blockIdx.x * 32;
    const int n0   = blockIdx.y * 256;
    const int kch  = K >> 5;

    v8f acc00 = (v8f){0.f,0.f,0.f,0.f,0.f,0.f,0.f,0.f};
    v8f acc01 = acc00, acc10 = acc00, acc11 = acc00;

    const int nsteps = K >> 6;
    stage_tiles(Xs, Ws, aF[0], bF[0], m0, n0, kch, 0, tid);

    for (int s = 0; s < nsteps; ++s) {
        const int buf = s & 1;
        ASYNC_DRAIN();
        __syncthreads();
        if (s + 1 < nsteps)
            stage_tiles(Xs, Ws, aF[buf ^ 1], bF[buf ^ 1],
                        m0, n0, kch, (s + 1) << 1, tid);

        const f16* aB = aF[buf];
        const f16* bp = bF[buf] + ((wave * 2) << 10) + (lane << 4);
        const v16h a00 = *(const v16h*)(aB + (lane << 4));
        const v16h a01 = *(const v16h*)(aB + 512  + (lane << 4));
        const v16h a10 = *(const v16h*)(aB + 1024 + (lane << 4));
        const v16h a11 = *(const v16h*)(aB + 1536 + (lane << 4));
        const v16h b00 = *(const v16h*)(bp);
        const v16h b01 = *(const v16h*)(bp + 512);
        const v16h b10 = *(const v16h*)(bp + 1024);
        const v16h b11 = *(const v16h*)(bp + 1536);

        acc00 = __builtin_amdgcn_wmma_f32_16x16x32_f16(false, a00, false, b00, (short)0, acc00, false, false);
        acc00 = __builtin_amdgcn_wmma_f32_16x16x32_f16(false, a01, false, b01, (short)0, acc00, false, false);
        acc01 = __builtin_amdgcn_wmma_f32_16x16x32_f16(false, a00, false, b10, (short)0, acc01, false, false);
        acc01 = __builtin_amdgcn_wmma_f32_16x16x32_f16(false, a01, false, b11, (short)0, acc01, false, false);
        acc10 = __builtin_amdgcn_wmma_f32_16x16x32_f16(false, a10, false, b00, (short)0, acc10, false, false);
        acc10 = __builtin_amdgcn_wmma_f32_16x16x32_f16(false, a11, false, b01, (short)0, acc10, false, false);
        acc11 = __builtin_amdgcn_wmma_f32_16x16x32_f16(false, a10, false, b10, (short)0, acc11, false, false);
        acc11 = __builtin_amdgcn_wmma_f32_16x16x32_f16(false, a11, false, b11, (short)0, acc11, false, false);
        __syncthreads();
    }

    const int half = lane >> 4;
    v8f accs[2][2] = {{acc00, acc01}, {acc10, acc11}};
#pragma unroll
    for (int mt = 0; mt < 2; ++mt) {
#pragma unroll
        for (int tl = 0; tl < 2; ++tl) {
            const int nn = n0 + (wave * 2 + tl) * 16 + (lane & 15);
            const float bb = bias ? bias[nn] : 0.f;
#pragma unroll
            for (int r = 0; r < 8; ++r) {
                const int mm = m0 + mt * 16 + r + (half << 3);
                float v = accs[mt][tl][r] + bb;
                if (act) v = 0.5f * v * (1.0f + erff(v * 0.70710678118654752f));
                if (Yf32) Yf32[(size_t)mm * N + nn] = v;
                if (Yf16) Yf16[swzA(mm, nn, N)] = (f16)v;
            }
        }
    }
}

// ---------------- RoPE + split into per-head [NH][T][HD] layout ----------------
__global__ __launch_bounds__(256)
void rope_split_kernel(const float* __restrict__ qkv, float* __restrict__ qh,
                       float* __restrict__ kh, float* __restrict__ vh) {
    int idx = blockIdx.x * blockDim.x + threadIdx.x;
    if (idx >= NH_SZ * T_SZ * 32) return;
    const int i = idx & 31;
    const int t = (idx >> 5) & (T_SZ - 1);
    const int h = idx >> 16;

    const float* row = qkv + (size_t)t * 3 * C_SZ;
    const float q1 = row[h * HD_SZ + i],            q2 = row[h * HD_SZ + i + 32];
    const float k1 = row[C_SZ + h * HD_SZ + i],     k2 = row[C_SZ + h * HD_SZ + i + 32];
    const float v1 = row[2 * C_SZ + h * HD_SZ + i], v2 = row[2 * C_SZ + h * HD_SZ + i + 32];

    const float inv = __expf(-(2.0f * (float)i / (float)HD_SZ) * 9.210340371976184f);
    float sn, cs;
    __sincosf((float)t * inv, &sn, &cs);

    const size_t base = ((size_t)h * T_SZ + t) * HD_SZ;
    qh[base + i]      = q1 * cs - q2 * sn;
    qh[base + i + 32] = q1 * sn + q2 * cs;
    kh[base + i]      = k1 * cs - k2 * sn;
    kh[base + i + 32] = k1 * sn + k2 * cs;
    vh[base + i]      = v1;
    vh[base + i + 32] = v2;
}

// ---------------- patch attention: one wave per (head, query) ----------------
__global__ __launch_bounds__(256)
void attn_kernel(const float* __restrict__ qh, const float* __restrict__ kh,
                 const float* __restrict__ vh, f16* __restrict__ outf,
                 int S, int step) {
    __shared__ float sc[8][512];
    const int wave = threadIdx.x >> 5;
    const int lane = threadIdx.x & 31;
    const int wid  = blockIdx.x * 8 + wave;
    const int h = wid / T_SZ;
    const int t = wid % T_SZ;

    int kbase, nk;
    if (t < S) { kbase = 0; nk = t + 1; }
    else {
        const int j = t - S;
        const int g = j / step;
        const int r = j % step;
        kbase = 1 + g * step;
        nk    = r + (S - step) + 1;
    }

    const float* qv = qh + ((size_t)h * T_SZ + t) * HD_SZ;
    const float scale = 0.125f;

    float lmax = -INFINITY;
    for (int c = lane; c < nk; c += 32) {
        const float* kr = kh + ((size_t)h * T_SZ + kbase + c) * HD_SZ;
        float d = 0.f;
#pragma unroll 8
        for (int e = 0; e < HD_SZ; ++e) d += qv[e] * kr[e];
        d *= scale;
        sc[wave][c] = d;
        lmax = fmaxf(lmax, d);
    }
    __syncthreads();
#pragma unroll
    for (int off = 16; off > 0; off >>= 1)
        lmax = fmaxf(lmax, __shfl_xor(lmax, off, 32));

    float lsum = 0.f;
    for (int c = lane; c < nk; c += 32) {
        float p = __expf(sc[wave][c] - lmax);
        sc[wave][c] = p;
        lsum += p;
    }
    __syncthreads();
#pragma unroll
    for (int off = 16; off > 0; off >>= 1)
        lsum += __shfl_xor(lsum, off, 32);
    const float rinv = 1.0f / lsum;

    float a0 = 0.f, a1 = 0.f;
    for (int c = 0; c < nk; ++c) {
        const float p = sc[wave][c];
        const float* vr = vh + ((size_t)h * T_SZ + kbase + c) * HD_SZ;
        a0 += p * vr[lane];
        a1 += p * vr[lane + 32];
    }
    outf[swzA(t, h * HD_SZ + lane,      C_SZ)] = (f16)(a0 * rinv);
    outf[swzA(t, h * HD_SZ + lane + 32, C_SZ)] = (f16)(a1 * rinv);
}

// ---------------- residual add ----------------
__global__ __launch_bounds__(256)
void add_kernel(float* __restrict__ x, const float* __restrict__ y, int n) {
    int i = blockIdx.x * blockDim.x + threadIdx.x;
    if (i < n) x[i] += y[i];
}

// ---------------- host-side orchestration ----------------
struct BlockW {
    const float *ff1_b, *ff1_w, *ff2_b, *ff2_w, *ln1_b, *ln1_g,
                *ln2_b, *ln2_g, *proj_b, *proj_w, *qkv_w;
};

extern "C" void kernel_launch(void* const* d_in, const int* in_sizes, int n_in,
                              void* d_out, int out_size, void* d_ws, size_t ws_size,
                              hipStream_t stream) {
    (void)in_sizes; (void)n_in; (void)out_size; (void)ws_size;

    const int* tokens = (const int*)d_in[0];
    auto F = [&](int i) { return (const float*)d_in[i]; };
    BlockW blk[3];
    for (int i = 0; i < 3; ++i) {
        const int s = 1 + 11 * i;
        blk[i] = { F(s + 0), F(s + 1), F(s + 2), F(s + 3), F(s + 4), F(s + 5),
                   F(s + 6), F(s + 7), F(s + 8), F(s + 9), F(s + 10) };
    }
    const float* embed  = F(34);
    const float* head_b = F(35);
    const float* head_w = F(36);
    const float* lnf_b  = F(37);
    const float* lnf_g  = F(38);

    float* x  = (float*)d_ws;
    float* t1 = x  + (size_t)T_SZ * C_SZ;
    float* qh = t1 + (size_t)T_SZ * 3 * C_SZ;
    float* kh = qh + (size_t)T_SZ * C_SZ;
    float* vh = kh + (size_t)T_SZ * C_SZ;
    f16* a16  = (f16*)(vh + (size_t)T_SZ * C_SZ);
    f16* h16  = a16 + (size_t)T_SZ * C_SZ;
    f16* w16  = h16 + (size_t)T_SZ * 4 * C_SZ;

    f16 *wq[3], *wp[3], *wf1[3], *wf2[3];
    size_t wo = 0;
    auto wconv = [&](const float* src, int N, int K) {
        f16* dst = w16 + wo;
        wo += (size_t)N * K;
        wconv_kernel<<<(N * K + 255) / 256, 256, 0, stream>>>(src, dst, N, K);
        return dst;
    };
    for (int i = 0; i < 3; ++i) {
        wq[i]  = wconv(blk[i].qkv_w,  3 * C_SZ, C_SZ);
        wp[i]  = wconv(blk[i].proj_w, C_SZ,     C_SZ);
        wf1[i] = wconv(blk[i].ff1_w,  4 * C_SZ, C_SZ);
        wf2[i] = wconv(blk[i].ff2_w,  C_SZ,     4 * C_SZ);
    }
    f16* whd = wconv(head_w, V_SZ, C_SZ);

    auto gemm = [&](const f16* X, const f16* W, const float* bias,
                    float* Y32, f16* Y16, int M, int N, int K, int act) {
        dim3 grid(M / 32, N / 256);
        gemm_xwt_wmma<<<grid, 256, 0, stream>>>(X, W, bias, Y32, Y16, M, N, K, act);
    };

    const int cfgS[3]    = {64, 512, 64};
    const int cfgStep[3] = {1, 256, 1};

    embed_kernel<<<(T_SZ * C_SZ) / 256, 256, 0, stream>>>(tokens, embed, x);

    for (int i = 0; i < 3; ++i) {
        const BlockW& p = blk[i];
        layernorm_kernel<<<T_SZ, 256, 0, stream>>>(x, p.ln1_g, p.ln1_b, a16);
        gemm(a16, wq[i], nullptr, t1, nullptr, T_SZ, 3 * C_SZ, C_SZ, 0);
        rope_split_kernel<<<(NH_SZ * T_SZ * 32) / 256, 256, 0, stream>>>(t1, qh, kh, vh);
        attn_kernel<<<(NH_SZ * T_SZ) / 8, 256, 0, stream>>>(qh, kh, vh, a16,
                                                            cfgS[i], cfgStep[i]);
        gemm(a16, wp[i], p.proj_b, t1, nullptr, T_SZ, C_SZ, C_SZ, 0);
        add_kernel<<<(T_SZ * C_SZ) / 256, 256, 0, stream>>>(x, t1, T_SZ * C_SZ);
        layernorm_kernel<<<T_SZ, 256, 0, stream>>>(x, p.ln2_g, p.ln2_b, a16);
        gemm(a16, wf1[i], p.ff1_b, nullptr, h16, T_SZ, 4 * C_SZ, C_SZ, 1);
        gemm(h16, wf2[i], p.ff2_b, t1, nullptr, T_SZ, C_SZ, 4 * C_SZ, 0);
        add_kernel<<<(T_SZ * C_SZ) / 256, 256, 0, stream>>>(x, t1, T_SZ * C_SZ);
    }

    layernorm_kernel<<<T_SZ, 256, 0, stream>>>(x, lnf_g, lnf_b, a16);
    gemm(a16, whd, head_b, (float*)d_out, nullptr, T_SZ, V_SZ, C_SZ, 0);
}
